// EquivariantGraphConv_7275674599802
// MI455X (gfx1250) — matrix operand order repaired and checked by reading, not compiled, not run on hardware
//
#include <hip/hip_runtime.h>
#include <hip/hip_bf16.h>
#include <math.h>

// ---------------------------------------------------------------------------
// EGNN layer for MI455X (gfx1250, wave32, WMMA 16x16x32 bf16, f32 accumulate)
// Software-pipelined operand fetch: B (global) and A (LDS) fragments are
// prefetched one step ahead of the consuming v_wmma.
// ---------------------------------------------------------------------------

typedef __bf16 bf16_t;
typedef __bf16 v16bf __attribute__((ext_vector_type(16)));
typedef __bf16 v8bf  __attribute__((ext_vector_type(8)));
typedef __bf16 v4bf  __attribute__((ext_vector_type(4)));
typedef float  v8f   __attribute__((ext_vector_type(8)));

#define N_NODES  20000
#define N_EDGES  640000
#define NODE_DIM 128
#define EDGE_DIM 64
#define HID      128

__device__ __forceinline__ float silu_f(float x) {
  return x / (1.0f + __expf(-x));
}

union ABfrag { v16bf v; v8bf h[2]; };

// D = A[16xK] * B[KxN] with K = 32*KTILES, N = 16*NTILES.
// A: bf16 in LDS, row-major, row stride `astride` elements.
//    A-frag layout used here (contiguous k, B packed to match): lane(l) -> m = l&15, kbase = (l>>4)*16;
//    elems 0..15 = K[kbase..kbase+15].
// B: pre-packed bf16 in global ws: record ((nt*KTILES+kb)*32 + lane)*16
//    holds 16 contiguous K values for column n = nt*16 + (lane&15),
//    k = kb*32 + (lane>>4)*16 + e.
// Fully unrolled with one-fragment-deep prefetch so the v_wmma never waits
// on the load it just issued.
template<int KTILES, int NTILES>
__device__ __forceinline__ void wmma_gemm16(
    const bf16_t* __restrict__ A, int astride,
    const bf16_t* __restrict__ Bp, int lane, v8f* acc)
{
  const int hi = lane >> 4, lo = lane & 15;
  const v8f zero = {0.f, 0.f, 0.f, 0.f, 0.f, 0.f, 0.f, 0.f};
#pragma unroll
  for (int nt = 0; nt < NTILES; ++nt) acc[nt] = zero;

  const bf16_t* brec = Bp + (size_t)lane * 16;           // + (nt*KTILES+kb)*512
  const bf16_t* arow = A + lo * astride + hi * 16;

  // Preload first A and B fragments.
  ABfrag a;
  a.h[0] = *(const v8bf*)(arow);
  a.h[1] = *(const v8bf*)(arow + 8);
  v16bf bcur = *(const v16bf*)(brec);                    // (nt=0, kb=0)

#pragma unroll
  for (int kb = 0; kb < KTILES; ++kb) {
    // Prefetch next kb's A fragment (LDS) while this kb computes.
    ABfrag anext;
    if (kb + 1 < KTILES) {
      anext.h[0] = *(const v8bf*)(arow + (kb + 1) * 32);
      anext.h[1] = *(const v8bf*)(arow + (kb + 1) * 32 + 8);
    }
#pragma unroll
    for (int nt = 0; nt < NTILES; ++nt) {
      // Prefetch next B fragment before consuming the current one.
      v16bf bnext = bcur;
      if (!(kb == KTILES - 1 && nt == NTILES - 1)) {
        const int nn = (nt == NTILES - 1) ? 0 : nt + 1;
        const int nk = (nt == NTILES - 1) ? kb + 1 : kb;
        bnext = *(const v16bf*)(brec + (size_t)(nn * KTILES + nk) * 512);
      }
      acc[nt] = __builtin_amdgcn_wmma_f32_16x16x32_bf16(
          false, a.v, false, bcur, (short)0, acc[nt], false, false);
      bcur = bnext;
    }
    if (kb + 1 < KTILES) { a.h[0] = anext.h[0]; a.h[1] = anext.h[1]; }
  }
}

// ---------------------------------------------------------------------------
// Weight pre-pack: f32 [K][N] row-major -> bf16 B-fragment records.
// ---------------------------------------------------------------------------
__global__ void pack_weight_kernel(const float* __restrict__ W, bf16_t* __restrict__ out,
                                   int Ktiles, int N, int total) {
  int p = blockIdx.x * blockDim.x + threadIdx.x;
  if (p >= total) return;
  int e    = p & 15;
  int lane = (p >> 4) & 31;
  int rest = p >> 9;
  int kb   = rest % Ktiles;
  int nt   = rest / Ktiles;
  int k = kb * 32 + (lane >> 4) * 16 + e;
  int n = nt * 16 + (lane & 15);
  out[p] = (bf16_t)W[k * N + n];
}

// Zero the aggregation buffer; seed out_coords with the input coords.
__global__ void init_kernel(const float* __restrict__ coords,
                            float* __restrict__ aggr,
                            float* __restrict__ out_coords) {
  int i = blockIdx.x * blockDim.x + threadIdx.x;
  if (i < N_NODES * NODE_DIM) aggr[i] = 0.0f;
  if (i < N_NODES * 3)        out_coords[i] = coords[i];
}

// ---------------------------------------------------------------------------
// Edge kernel: one wave per 16-edge tile, 4 waves (4 tiles) per workgroup.
//   edge_feat[16x320] -> silu(.@eW1+b) -> silu(.@eW2+b) = msg[16x128]
//   atomicAdd msg into aggr[dst]; coord gate silu(msg@cW1+b)@cW2+b scattered.
// ---------------------------------------------------------------------------
__global__ __launch_bounds__(128) void egnn_edge_kernel(
    const float* __restrict__ node_feat,
    const int*   __restrict__ edge_index,
    const float* __restrict__ edge_attr,
    const float* __restrict__ coords,
    const float* __restrict__ eb1,
    const float* __restrict__ eb2,
    const float* __restrict__ cb1,
    const float* __restrict__ cW2,
    const float* __restrict__ cb2,
    const bf16_t* __restrict__ eW1p,
    const bf16_t* __restrict__ eW2p,
    const bf16_t* __restrict__ cW1p,
    float* __restrict__ aggr,
    float* __restrict__ out_coords)
{
  // Per-wave LDS slice, buffers overlapped in time:
  //   [0     .. 10239] X  [16][320] bf16   (live: gather -> layer1)
  //   [0     ..  4095] M  [16][128] bf16   (live: layer2 -> coord gemm)
  //   [4096  ..  8191] G  [16][64]  f32    (live: coord gemm -> dot)
  //   [10240 .. 14335] H  [16][128] bf16   (live: layer1 -> layer2)
  __shared__ __align__(16) char smem[4][14336];
  __shared__ int SrcS[4][16];
  __shared__ int DstS[4][16];

  const int w    = threadIdx.x >> 5;
  const int lane = threadIdx.x & 31;
  const int hi = lane >> 4, lo = lane & 15;
  const int tile = blockIdx.x * 4 + w;
  const int e0 = tile * 16;

  bf16_t* Xs = (bf16_t*)(&smem[w][0]);
  bf16_t* Hs = (bf16_t*)(&smem[w][10240]);
  bf16_t* Ms = (bf16_t*)(&smem[w][0]);
  float*  Gs = (float*)(&smem[w][4096]);

  if (lane < 16) {
    SrcS[w][lane] = edge_index[e0 + lane];
    DstS[w][lane] = edge_index[N_EDGES + e0 + lane];
  }

  // Gather + convert to bf16: 16 edges x 320 feats = 1280 float4 chunks / 16.
  const float4* nf4 = (const float4*)node_feat;
  const float4* ea4 = (const float4*)edge_attr;
  for (int i = lane; i < 16 * 80; i += 32) {
    int e = i / 80;
    int c = i - e * 80;
    float4 v;
    if (c < 32)      v = nf4[SrcS[w][e] * 32 + c];
    else if (c < 64) v = nf4[DstS[w][e] * 32 + (c - 32)];
    else             v = ea4[(e0 + e) * 16 + (c - 64)];
    v4bf pk = { (bf16_t)v.x, (bf16_t)v.y, (bf16_t)v.z, (bf16_t)v.w };
    *(v4bf*)(Xs + e * 320 + c * 4) = pk;
  }

  v8f acc[8];

  // ---- edge MLP layer 1: [16x320] @ [320x128] ----
  wmma_gemm16<10, 8>(Xs, 320, eW1p, lane, acc);
#pragma unroll
  for (int nt = 0; nt < 8; ++nt) {
    float b = eb1[nt * 16 + lo];
#pragma unroll
    for (int r = 0; r < 8; ++r) {
      float x = silu_f(acc[nt][r] + b);
      Hs[(r + 8 * hi) * 128 + nt * 16 + lo] = (bf16_t)x;   // D: m=r+8*hi, n=nt*16+lo
    }
  }

  // ---- edge MLP layer 2: [16x128] @ [128x128], scatter-add to aggr ----
  wmma_gemm16<4, 8>(Hs, 128, eW2p, lane, acc);
  int dn[8];
#pragma unroll
  for (int r = 0; r < 8; ++r) dn[r] = DstS[w][r + 8 * hi];
#pragma unroll
  for (int nt = 0; nt < 8; ++nt) {
    float b = eb2[nt * 16 + lo];
#pragma unroll
    for (int r = 0; r < 8; ++r) {
      float x = silu_f(acc[nt][r] + b);
      Ms[(r + 8 * hi) * 128 + nt * 16 + lo] = (bf16_t)x;
      atomicAdd(&aggr[dn[r] * 128 + nt * 16 + lo], x);
    }
  }

  // ---- coord gate: G = silu(M @ cW1 + cb1), [16x128]@[128x64] ----
  v8f g[4];
  wmma_gemm16<4, 4>(Ms, 128, cW1p, lane, g);
#pragma unroll
  for (int nt = 0; nt < 4; ++nt) {
    float b = cb1[nt * 16 + lo];
#pragma unroll
    for (int r = 0; r < 8; ++r)
      Gs[(r + 8 * hi) * 64 + nt * 16 + lo] = silu_f(g[nt][r] + b);
  }

  // per-edge scalar weight: w = G[e] . cW2 + cb2 (lane pairs split the K=64 dot)
  float p = 0.f;
#pragma unroll
  for (int j = 0; j < 32; ++j)
    p += Gs[lo * 64 + hi * 32 + j] * cW2[hi * 32 + j];
  float q = p + __shfl_xor(p, 16, 32);

  if (lane < 16) {
    float wgt = q + cb2[0];
    int s = SrcS[w][lane], d = DstS[w][lane];
    float dx = coords[s * 3 + 0] - coords[d * 3 + 0];
    float dy = coords[s * 3 + 1] - coords[d * 3 + 1];
    float dz = coords[s * 3 + 2] - coords[d * 3 + 2];
    float inv = 1.0f / (sqrtf(dx * dx + dy * dy + dz * dz) + 1e-8f);
    atomicAdd(&out_coords[d * 3 + 0], wgt * dx * inv);
    atomicAdd(&out_coords[d * 3 + 1], wgt * dy * inv);
    atomicAdd(&out_coords[d * 3 + 2], wgt * dz * inv);
  }
}

// ---------------------------------------------------------------------------
// Node kernel: residual node MLP, one wave per 16-node tile.
// ---------------------------------------------------------------------------
__global__ __launch_bounds__(128) void egnn_node_kernel(
    const float* __restrict__ node_feat,
    const float* __restrict__ aggr,
    const float* __restrict__ nb1,
    const float* __restrict__ nb2,
    const bf16_t* __restrict__ nW1p,
    const bf16_t* __restrict__ nW2p,
    float* __restrict__ out_nodes)
{
  __shared__ __align__(16) bf16_t Ys[4][16][256];
  __shared__ __align__(16) bf16_t Hs[4][16][128];
  const int w    = threadIdx.x >> 5;
  const int lane = threadIdx.x & 31;
  const int hi = lane >> 4, lo = lane & 15;
  const int tile = blockIdx.x * 4 + w;
  if (tile >= N_NODES / 16) return;      // wave-uniform exit
  const int n0 = tile * 16;

  const float4* nf4 = (const float4*)node_feat;
  const float4* ag4 = (const float4*)aggr;
  for (int i = lane; i < 16 * 64; i += 32) {
    int e = i >> 6;
    int c = i & 63;
    float4 v = (c < 32) ? nf4[(n0 + e) * 32 + c] : ag4[(n0 + e) * 32 + (c - 32)];
    v4bf pk = { (bf16_t)v.x, (bf16_t)v.y, (bf16_t)v.z, (bf16_t)v.w };
    *(v4bf*)(&Ys[w][e][c * 4]) = pk;
  }

  v8f acc[8];
  wmma_gemm16<8, 8>(&Ys[w][0][0], 256, nW1p, lane, acc);
#pragma unroll
  for (int nt = 0; nt < 8; ++nt) {
    float b = nb1[nt * 16 + lo];
#pragma unroll
    for (int r = 0; r < 8; ++r)
      Hs[w][r + 8 * hi][nt * 16 + lo] = (bf16_t)silu_f(acc[nt][r] + b);
  }
  wmma_gemm16<4, 8>(&Hs[w][0][0], 128, nW2p, lane, acc);
#pragma unroll
  for (int nt = 0; nt < 8; ++nt) {
    float b = nb2[nt * 16 + lo];
#pragma unroll
    for (int r = 0; r < 8; ++r) {
      int nd  = n0 + r + 8 * hi;
      int col = nt * 16 + lo;
      out_nodes[nd * 128 + col] = node_feat[nd * 128 + col] + acc[nt][r] + b;
    }
  }
}

// ---------------------------------------------------------------------------
extern "C" void kernel_launch(void* const* d_in, const int* in_sizes, int n_in,
                              void* d_out, int out_size, void* d_ws, size_t ws_size,
                              hipStream_t stream) {
  const float* node_feat  = (const float*)d_in[0];
  const int*   edge_index = (const int*)d_in[1];
  const float* edge_attr  = (const float*)d_in[2];
  const float* coords     = (const float*)d_in[3];
  const float* eW1 = (const float*)d_in[4];
  const float* eb1 = (const float*)d_in[5];
  const float* eW2 = (const float*)d_in[6];
  const float* eb2 = (const float*)d_in[7];
  const float* nW1 = (const float*)d_in[8];
  const float* nb1 = (const float*)d_in[9];
  const float* nW2 = (const float*)d_in[10];
  const float* nb2 = (const float*)d_in[11];
  const float* cW1 = (const float*)d_in[12];
  const float* cb1 = (const float*)d_in[13];
  const float* cW2 = (const float*)d_in[14];
  const float* cb2 = (const float*)d_in[15];

  float* out_nodes  = (float*)d_out;
  float* out_coords = out_nodes + (size_t)N_NODES * NODE_DIM;

  // Workspace carve-out (~10.5 MB total).
  char* ws = (char*)d_ws;
  float*  aggr = (float*)ws;    ws += (size_t)N_NODES * NODE_DIM * sizeof(float);
  bf16_t* eW1p = (bf16_t*)ws;   ws += (size_t)320 * 128 * sizeof(bf16_t);
  bf16_t* eW2p = (bf16_t*)ws;   ws += (size_t)128 * 128 * sizeof(bf16_t);
  bf16_t* nW1p = (bf16_t*)ws;   ws += (size_t)256 * 128 * sizeof(bf16_t);
  bf16_t* nW2p = (bf16_t*)ws;   ws += (size_t)128 * 128 * sizeof(bf16_t);
  bf16_t* cW1p = (bf16_t*)ws;   ws += (size_t)128 * 64  * sizeof(bf16_t);

  pack_weight_kernel<<<(320 * 128 + 255) / 256, 256, 0, stream>>>(eW1, eW1p, 10, 128, 320 * 128);
  pack_weight_kernel<<<(128 * 128 + 255) / 256, 256, 0, stream>>>(eW2, eW2p, 4, 128, 128 * 128);
  pack_weight_kernel<<<(256 * 128 + 255) / 256, 256, 0, stream>>>(nW1, nW1p, 8, 128, 256 * 128);
  pack_weight_kernel<<<(128 * 128 + 255) / 256, 256, 0, stream>>>(nW2, nW2p, 4, 128, 128 * 128);
  pack_weight_kernel<<<(128 * 64  + 255) / 256, 256, 0, stream>>>(cW1, cW1p, 4, 64, 128 * 64);

  init_kernel<<<(N_NODES * NODE_DIM + 255) / 256, 256, 0, stream>>>(coords, aggr, out_coords);

  egnn_edge_kernel<<<N_EDGES / 16 / 4, 128, 0, stream>>>(
      node_feat, edge_index, edge_attr, coords,
      eb1, eb2, cb1, cW2, cb2, eW1p, eW2p, cW1p, aggr, out_coords);

  egnn_node_kernel<<<(N_NODES / 16 + 3) / 4, 128, 0, stream>>>(
      node_feat, aggr, nb1, nb2, nW1p, nW2p, out_nodes);
}
